// CINs_59485297050176
// MI455X (gfx1250) — matrix operand order repaired and checked by reading, not compiled, not run on hardware
//
#include <hip/hip_runtime.h>

// CIN layers as on-the-fly-generated GEMMs on the WMMA f16 path (f32 accum).
// Rows r = (b,d), R = 16384. Per layer: Y[r,h] = sum_i xprev[r,i] * (x0[r,:] . W[i,:,h])
// For fixed i the A tile is (per-row scalar) * (constant x0 tile): x0 fragments live
// in registers; per-i A update is v_pk_mul_f16. W slices are pre-converted to f16 in
// the exact LDS layout by a prep kernel and streamed in with double-buffered
// global_load_async_to_lds_b128 (ASYNCcnt), overlapping the WMMAs.

typedef __attribute__((ext_vector_type(16))) _Float16 v16h;
typedef __attribute__((ext_vector_type(8)))  float    v8f;
typedef __attribute__((ext_vector_type(4)))  unsigned int v4u;

#define F0       39
#define DD       16
#define HH       128
#define KPAD     64               // F0 padded to 2 x K=32 WMMA chunks
#define MROWS    64               // rows (b,d) per workgroup
#define NTHREADS 256              // 8 wave32: 2 row-halves x 4 col-quarters
#define RTOT     16384            // B * D
#define SLICE    (HH * KPAD)      // f16 elems per prepared W slice (16 KB)

union F16Frag { v4u q[2]; v16h v; };

__device__ __forceinline__ void async_ld16(unsigned lds_off, const void* sbase,
                                           unsigned byte_off) {
    // GVS mode: LDS[lds_off] = MEM[sbase + byte_off], 16 bytes, ASYNCcnt-tracked
    asm volatile("global_load_async_to_lds_b128 %0, %1, %2"
                 :: "v"(lds_off), "v"(byte_off), "s"(sbase) : "memory");
}
__device__ __forceinline__ void wait_async0() {
    asm volatile("s_wait_asynccnt 0" ::: "memory");
}

// Pre-format W: Wt[i*SLICE + h*KPAD + j] = (f16) W[(i*F0+j)*H + h], pad j>=F0 -> 0
__global__ __launch_bounds__(256)
void prep_w(const float* __restrict__ W, _Float16* __restrict__ Wt, int total) {
    int t = blockIdx.x * blockDim.x + threadIdx.x;
    if (t >= total) return;
    int j = t & (KPAD - 1);
    int h = (t >> 6) & (HH - 1);
    int i = t >> 13;
    Wt[t] = (j < F0) ? (_Float16)W[(i * F0 + j) * HH + h] : (_Float16)0.f;
}

template <int P, bool PREV_F32_INPUT>
__global__ __launch_bounds__(NTHREADS)
void cin_layer(const float*    __restrict__ x0,    // [B, F0, D] f32
               const void*     __restrict__ xprev, // layer0: x0 (f32); else f16 [R, P]
               const _Float16* __restrict__ Wt,    // [P, H, KPAD] f16 (prepped)
               _Float16*       __restrict__ yout,  // [R, H] f16
               float*          __restrict__ pooled,// [B, 3*H] f32
               int layerOfs)
{
    __shared__ _Float16 sX0[MROWS * KPAD];   // x0 rows, zero-padded
    __shared__ _Float16 sXp[MROWS * P];      // prev-activation rows (scales)
    __shared__ _Float16 sW [2][SLICE];       // double-buffered W slice [h][j]

    const int tid  = threadIdx.x;
    const int lane = tid & 31;
    const int wave = tid >> 5;
    const int wr   = wave >> 2;     // 0..1 : 32-row half
    const int wc   = wave & 3;      // 0..3 : 32-col quarter
    const int hi   = lane >> 4;
    const int m    = lane & 15;
    const int rowBase = blockIdx.x * MROWS;

    const unsigned sW0off = (unsigned)(size_t)&sW[0][0];
    const unsigned sW1off = (unsigned)(size_t)&sW[1][0];
    const unsigned tOff   = (unsigned)tid * 64u;   // 64 B per thread per 16 KB block

    // ---- async: prev activations (f16 path) are one contiguous 16 KB block ----
    if (!PREV_F32_INPUT) {
        const char* src = (const char*)xprev + (size_t)rowBase * P * sizeof(_Float16);
        const unsigned sXpoff = (unsigned)(size_t)&sXp[0];
#pragma unroll
        for (int c = 0; c < 4; ++c)
            async_ld16(sXpoff + tOff + c * 16u, src, tOff + c * 16u);
    }
    // ---- async: W slice 0 into buffer 0 ----
#pragma unroll
    for (int c = 0; c < 4; ++c)
        async_ld16(sW0off + tOff + c * 16u, (const void*)Wt, tOff + c * 16u);

    // ---- gather x0 rows as f16, zero-padded j in [F0, KPAD) ----
    for (int t = tid; t < MROWS * KPAD; t += NTHREADS) {
        int row = t >> 6, j = t & 63;
        int r = rowBase + row, b = r >> 4, d = r & 15;
        float v = (j < F0) ? x0[(b * F0 + j) * DD + d] : 0.f;
        sX0[row * KPAD + j] = (_Float16)v;
    }
    // ---- layer0: gather strided f32 scales ----
    if (PREV_F32_INPUT) {
        for (int t = tid; t < MROWS * P; t += NTHREADS) {
            int row = t / P, i = t - row * P;
            int r = rowBase + row, b = r >> 4, d = r & 15;
            sXp[row * P + i] = (_Float16)((const float*)xprev)[(b * P + i) * DD + d];
        }
    }
    wait_async0();
    __syncthreads();

    // ---- x0 base A-fragments, registers for the whole kernel ----
    // A 16x32 f16 layout: lane(m,hi): elems 0..7 -> K = kc*32 + hi*8 + e,
    //                                 elems 8..15 -> K = kc*32 + 16 + hi*8 + e
    F16Frag A0[2][2]; // [Mt][kc]
#pragma unroll
    for (int Mt = 0; Mt < 2; ++Mt)
#pragma unroll
        for (int kc = 0; kc < 2; ++kc) {
            const _Float16* p = &sX0[(wr * 32 + Mt * 16 + m) * KPAD + kc * 32 + hi * 8];
            A0[Mt][kc].q[0] = *(const v4u*)p;
            A0[Mt][kc].q[1] = *(const v4u*)(p + 16);
        }

    v8f acc[2][2] = {}; // [Mt][Nt]

    for (int i = 0; i < P; ++i) {
        // issue async copy of slice i+1 into the other buffer (overlaps compute)
        if (i + 1 < P) {
            const void* src = (const void*)(Wt + (size_t)(i + 1) * SLICE);
            const unsigned dstoff = ((i + 1) & 1) ? sW1off : sW0off;
#pragma unroll
            for (int c = 0; c < 4; ++c)
                async_ld16(dstoff + tOff + c * 16u, src, tOff + c * 16u);
        }

        const _Float16* sWi = &sW[i & 1][0];

        // per-row scale factors (row depends only on m -> uniform across hi)
        _Float16 a0 = sXp[(wr * 32 +  0 + m) * P + i];
        _Float16 a1 = sXp[(wr * 32 + 16 + m) * P + i];

        // B fragments: B 32x16 f16 layout -> same K order as A (hi*8 + e, 16 + hi*8 + e)
        F16Frag Bf[2][2]; // [kc][Nt]
#pragma unroll
        for (int kc = 0; kc < 2; ++kc)
#pragma unroll
            for (int Nt = 0; Nt < 2; ++Nt) {
                const _Float16* p = &sWi[(wc * 32 + Nt * 16 + m) * KPAD + kc * 32 + hi * 8];
                Bf[kc][Nt].q[0] = *(const v4u*)p;
                Bf[kc][Nt].q[1] = *(const v4u*)(p + 16);
            }

        // scale A fragments: 8 x v_pk_mul_f16 each
        v16h As[2][2];
#pragma unroll
        for (int kc = 0; kc < 2; ++kc) {
            As[0][kc] = A0[0][kc].v * a0;
            As[1][kc] = A0[1][kc].v * a1;
        }

        // 8 WMMAs per i
#pragma unroll
        for (int Mt = 0; Mt < 2; ++Mt)
#pragma unroll
            for (int Nt = 0; Nt < 2; ++Nt)
#pragma unroll
                for (int kc = 0; kc < 2; ++kc)
                    acc[Mt][Nt] = __builtin_amdgcn_wmma_f32_16x16x32_f16(
                        false, As[Mt][kc], false, Bf[kc][Nt].v,
                        (short)0, acc[Mt][Nt], false, false);

        // next iteration needs slice i+1 resident and everyone done with it
        if (i + 1 < P) {
            wait_async0();
            __syncthreads();
        }
    }

    // ---- writeback: Y (f16, next layer) + pooled sums over d ----
    // C/D layout: lane owns column n=m, rows M = 8*hi + g
#pragma unroll
    for (int Mt = 0; Mt < 2; ++Mt) {
        const int rbaseM = rowBase + wr * 32 + Mt * 16; // 16 rows == 16 d of one b
#pragma unroll
        for (int Nt = 0; Nt < 2; ++Nt) {
            const int h = wc * 32 + Nt * 16 + m;
            float s = 0.f;
#pragma unroll
            for (int g = 0; g < 8; ++g) {
                const int row = rbaseM + hi * 8 + g;
                yout[row * HH + h] = (_Float16)acc[Mt][Nt][g];
                s += acc[Mt][Nt][g];
            }
            s += __shfl_xor(s, 16, 32);
            if (hi == 0) {
                const int b = rbaseM >> 4;
                pooled[b * (3 * HH) + layerOfs + h] = s;
            }
        }
    }
}

extern "C" void kernel_launch(void* const* d_in, const int* in_sizes, int n_in,
                              void* d_out, int out_size, void* d_ws, size_t ws_size,
                              hipStream_t stream) {
    const float* x0 = (const float*)d_in[0]; // [1024, 39, 16]
    const float* W0 = (const float*)d_in[1]; // [1521, 128]
    const float* W1 = (const float*)d_in[2]; // [4992, 128]
    const float* W2 = (const float*)d_in[3]; // [4992, 128]
    float* out = (float*)d_out;              // [1024, 384]

    _Float16* actA = (_Float16*)d_ws;                  // [R, H] f16
    _Float16* actB = actA + (size_t)RTOT * HH;         // [R, H] f16
    _Float16* Wt0  = actB + (size_t)RTOT * HH;         // [39,  H, KPAD] f16
    _Float16* Wt1  = Wt0 + (size_t)F0 * SLICE;         // [128, H, KPAD] f16
    _Float16* Wt2  = Wt1 + (size_t)HH * SLICE;         // [128, H, KPAD] f16

    int tot0 = F0 * SLICE, tot12 = HH * SLICE;
    prep_w<<<(tot0  + 255) / 256, 256, 0, stream>>>(W0, Wt0, tot0);
    prep_w<<<(tot12 + 255) / 256, 256, 0, stream>>>(W1, Wt1, tot12);
    prep_w<<<(tot12 + 255) / 256, 256, 0, stream>>>(W2, Wt2, tot12);

    dim3 grid(RTOT / MROWS), block(NTHREADS);
    cin_layer<F0, true ><<<grid, block, 0, stream>>>(x0, (const void*)x0,   Wt0, actA, out, 0);
    cin_layer<HH, false><<<grid, block, 0, stream>>>(x0, (const void*)actA, Wt1, actB, out, HH);
    cin_layer<HH, false><<<grid, block, 0, stream>>>(x0, (const void*)actB, Wt2, actA, out, 2 * HH);
}
